// GINet_39470749450579
// MI455X (gfx1250) — compile-verified, matrix operand order checked by reading, not run on hardware
//
#include <hip/hip_runtime.h>
#include <hip/hip_bf16.h>

#define NN 131072
#define EE 262144
#define GG 4096
#define DD 300
#define DP 320
#define LL 5
#define HH 600
#define HP 640
#define FEATN 512
#define EXN 64
#define TIN 768
#define PDN 288
#define CZ 576
#define BNEPS 1e-5f

typedef __attribute__((ext_vector_type(16))) __bf16 v16bf;
typedef __attribute__((ext_vector_type(8)))  float  v8f;

union FragBF { v16bf v; unsigned u[8]; uint4 q[2]; };
union FragF  { v8f  v; float    f[8]; };

__device__ __forceinline__ unsigned short f2bf(float f) {
  unsigned u = __float_as_uint(f);
  u += 0x7FFFu + ((u >> 16) & 1u);          // round-to-nearest-even
  return (unsigned short)(u >> 16);
}
__device__ __forceinline__ float bf2f(unsigned short h) {
  return __uint_as_float(((unsigned)h) << 16);
}
__device__ __forceinline__ unsigned pack2(float a, float b) {
  return (unsigned)f2bf(a) | ((unsigned)f2bf(b) << 16);
}
__device__ __forceinline__ float softplusf(float x) {
  return (x > 20.f) ? x : log1pf(__expf(x));
}
__device__ __forceinline__ void atomAddF(float* p, float v) {
  __hip_atomic_fetch_add(p, v, __ATOMIC_RELAXED, __HIP_MEMORY_SCOPE_AGENT);
}

// ---------------------------------------------------------------------------
// Generic bf16 WMMA GEMM: C[M,Nc] = act(A[M,K] * Bt[Nc,K]^T + bias)
// Block: 256 thr = 8 waves; block tile 256 x 32; wave tile 32 x 32
// (2 row-frags x 2 col-frags = 4 x v_wmma_f32_16x16x32_bf16 per K-step).
// A row-major bf16 (lda elems), Bt row-major [Nc][K] bf16 (ldb = K elems).
// B tile staged to LDS with GLOBAL_LOAD_ASYNC_TO_LDS_B64 (ASYNCcnt).
// M % 256 == 0, Nc % 32 == 0, K % 32 == 0, lda % 8 == 0 (caller guarantees).
// ACT: 0 none, 1 relu, 2 softplus.
// ---------------------------------------------------------------------------
template<int ACT>
__global__ void __launch_bounds__(256)
k_gemm(const unsigned short* __restrict__ A, int lda,
       const unsigned short* __restrict__ Bt, int ldb,
       const float* __restrict__ bias,
       float* __restrict__ Cf, unsigned short* __restrict__ Cb, int ldc,
       int K)
{
  // [col][k dword]; 20-dword row stride: rows 16B-aligned, 20*c mod 64 distinct
  // for c=0..15 -> conflict-free ds_load_b128 fragment reads.
  __shared__ __align__(16) unsigned sB[32][20];

  const int tid  = threadIdx.x;
  const int wave = tid >> 5;
  const int lane = tid & 31;
  const int l16  = lane & 15;
  const int half = lane >> 4;
  const int rowBase = blockIdx.y * 256 + wave * 32;
  const int colBase = blockIdx.x * 32;

  FragF acc00, acc01, acc10, acc11;
#pragma unroll
  for (int i = 0; i < 8; ++i) {
    acc00.f[i] = 0.f; acc01.f[i] = 0.f; acc10.f[i] = 0.f; acc11.f[i] = 0.f;
  }

  const unsigned* B32 = reinterpret_cast<const unsigned*>(Bt);
  const uint4*    A4  = reinterpret_cast<const uint4*>(A);
  const uint4*    S4  = reinterpret_cast<const uint4*>(&sB[0][0]); // row = 5 uint4
  const int lda4 = lda >> 3;
  const int ldb2 = ldb >> 1;

  // Per-thread async-staging addresses: thread moves one b64 (2 dwords) per tile.
  const int sc  = (tid * 2) >> 4;                 // 0..31  (B column in tile)
  const int skp = (tid * 2) & 15;                 // even   (k-dword in tile)
  const unsigned ldsDst = (unsigned)(unsigned long long)&sB[sc][skp];
  const unsigned* gsrc  = B32 + (size_t)(colBase + sc) * ldb2 + skp;

  const int r0 = rowBase + l16;
  const int r1 = rowBase + 16 + l16;

  for (int k0 = 0; k0 < K; k0 += 32) {
    __syncthreads();                               // previous tile fully consumed
    {
      unsigned long long ga = (unsigned long long)(const char*)(gsrc + (k0 >> 1));
      asm volatile("global_load_async_to_lds_b64 %0, %1, off"
                   :: "v"(ldsDst), "v"(ga) : "memory");
      asm volatile("s_wait_asynccnt 0x0" ::: "memory");
    }
    __syncthreads();                               // all waves' tile writes visible

    const int kb4 = k0 >> 3;                       // uint4 offset along K
    FragBF a0, a1, b0, b1;
    a0.q[0] = A4[r0 * lda4 + kb4 + half];
    a0.q[1] = A4[r0 * lda4 + kb4 + 2 + half];
    a1.q[0] = A4[r1 * lda4 + kb4 + half];
    a1.q[1] = A4[r1 * lda4 + kb4 + 2 + half];
    b0.q[0] = S4[l16 * 5 + half];
    b0.q[1] = S4[l16 * 5 + 2 + half];
    b1.q[0] = S4[(16 + l16) * 5 + half];
    b1.q[1] = S4[(16 + l16) * 5 + 2 + half];

    acc00.v = __builtin_amdgcn_wmma_f32_16x16x32_bf16(false, a0.v, false, b0.v,
                                                      (short)0, acc00.v, false, false);
    acc01.v = __builtin_amdgcn_wmma_f32_16x16x32_bf16(false, a0.v, false, b1.v,
                                                      (short)0, acc01.v, false, false);
    acc10.v = __builtin_amdgcn_wmma_f32_16x16x32_bf16(false, a1.v, false, b0.v,
                                                      (short)0, acc10.v, false, false);
    acc11.v = __builtin_amdgcn_wmma_f32_16x16x32_bf16(false, a1.v, false, b1.v,
                                                      (short)0, acc11.v, false, false);

    if (k0 + 32 < K) {                             // prefetch next A tile
      __builtin_prefetch(&A4[r0 * lda4 + kb4 + 4], 0, 1);
      __builtin_prefetch(&A4[r1 * lda4 + kb4 + 4], 0, 1);
    }
  }

  const float bia0 = bias ? bias[colBase + l16]      : 0.f;
  const float bia1 = bias ? bias[colBase + 16 + l16] : 0.f;
#pragma unroll
  for (int r = 0; r < 8; ++r) {
    int rowA = rowBase + half * 8 + r;
    int rowB = rowBase + 16 + half * 8 + r;
    float v00 = acc00.f[r] + bia0, v01 = acc01.f[r] + bia1;
    float v10 = acc10.f[r] + bia0, v11 = acc11.f[r] + bia1;
    if (ACT == 1) {
      v00 = fmaxf(v00, 0.f); v01 = fmaxf(v01, 0.f);
      v10 = fmaxf(v10, 0.f); v11 = fmaxf(v11, 0.f);
    }
    if (ACT == 2) {
      v00 = softplusf(v00); v01 = softplusf(v01);
      v10 = softplusf(v10); v11 = softplusf(v11);
    }
    if (Cf) {
      Cf[rowA * ldc + colBase + l16]      = v00;
      Cf[rowA * ldc + colBase + 16 + l16] = v01;
      Cf[rowB * ldc + colBase + l16]      = v10;
      Cf[rowB * ldc + colBase + 16 + l16] = v11;
    }
    if (Cb) {
      Cb[rowA * ldc + colBase + l16]      = f2bf(v00);
      Cb[rowA * ldc + colBase + 16 + l16] = f2bf(v01);
      Cb[rowB * ldc + colBase + l16]      = f2bf(v10);
      Cb[rowB * ldc + colBase + 16 + l16] = f2bf(v11);
    }
  }
}

// --------------------------- small utility kernels -------------------------

__global__ void k_zero(float* p, int n) {
  int i = blockIdx.x * blockDim.x + threadIdx.x;
  if (i < n) p[i] = 0.f;
}

// f32 -> bf16, pair-packed
__global__ void k_cvt_pairs(const float* __restrict__ src, unsigned* __restrict__ dst, int npairs) {
  int i = blockIdx.x * blockDim.x + threadIdx.x;
  if (i >= npairs) return;
  float2 s = reinterpret_cast<const float2*>(src)[i];
  dst[i] = pack2(s.x, s.y);
}

// transpose f32 [Kv,Nv] -> bf16 [Np][Kp] zero-padded
__global__ void k_wconv_t(const float* __restrict__ src, unsigned short* __restrict__ dst,
                          int Kv, int Nv, int Kp, int Np) {
  int i = blockIdx.x * blockDim.x + threadIdx.x;
  if (i >= Np * Kp) return;
  int n = i / Kp, k = i % Kp;
  float v = (k < Kv && n < Nv) ? src[k * Nv + n] : 0.f;
  dst[n * Kp + k] = f2bf(v);
}

__global__ void k_biaspad(const float* __restrict__ src, float* __restrict__ dst,
                          int nvalid, int ntot) {
  int i = blockIdx.x * blockDim.x + threadIdx.x;
  if (i < ntot) dst[i] = (i < nvalid) ? src[i] : 0.f;
}

// h = x_emb1[x0] + x_emb2[x1], bf16 pairs, zero-padded to DP
__global__ void k_embed(const int* __restrict__ x,
                        const float* __restrict__ e1, const float* __restrict__ e2,
                        unsigned* __restrict__ hbf) {
  int i = blockIdx.x * blockDim.x + threadIdx.x;   // NN * DP/2
  if (i >= NN * (DP / 2)) return;
  int n = i / (DP / 2), j = i % (DP / 2), c0 = j * 2;
  if (c0 >= DD) { hbf[i] = 0u; return; }
  int x0 = x[n * 2], x1 = x[n * 2 + 1];
  float v0 = e1[x0 * DD + c0]     + e2[x1 * DD + c0];
  float v1 = e1[x0 * DD + c0 + 1] + e2[x1 * DD + c0 + 1];
  hbf[i] = pack2(v0, v1);
}

// agg = h + selfloop_emb (f32), pad cols zeroed
__global__ void k_agg_init(const unsigned* __restrict__ hbf,
                           const float* __restrict__ se1, const float* __restrict__ se2,
                           float* __restrict__ agg) {
  int i = blockIdx.x * blockDim.x + threadIdx.x;   // NN * DP/2
  if (i >= NN * (DP / 2)) return;
  int j = i % (DP / 2), c0 = j * 2;
  float2 out;
  if (c0 >= DD) { out.x = 0.f; out.y = 0.f; }
  else {
    unsigned p = hbf[i];
    out.x = bf2f((unsigned short)(p & 0xffffu)) + se1[c0]     + se2[c0];
    out.y = bf2f((unsigned short)(p >> 16))     + se1[c0 + 1] + se2[c0 + 1];
  }
  reinterpret_cast<float2*>(agg)[i] = out;
}

// scatter-add edge messages: 4 channels per thread
__global__ void k_edge(const int* __restrict__ ei, const int* __restrict__ ea,
                       const unsigned* __restrict__ hbf,
                       const float* __restrict__ ge1, const float* __restrict__ ge2,
                       float* __restrict__ agg) {
  int i = blockIdx.x * blockDim.x + threadIdx.x;   // EE * 75
  if (i >= EE * 75) return;
  int e = i / 75, j = i % 75, c0 = j * 4;
  int s  = ei[e], d = ei[EE + e];
  int a1 = ea[e * 2], a2 = ea[e * 2 + 1];
  const float* p1 = ge1 + a1 * DD + c0;
  const float* p2 = ge2 + a2 * DD + c0;
  unsigned q0 = hbf[s * (DP / 2) + (c0 >> 1)];
  unsigned q1 = hbf[s * (DP / 2) + (c0 >> 1) + 1];
  float h0 = bf2f((unsigned short)(q0 & 0xffffu));
  float h1 = bf2f((unsigned short)(q0 >> 16));
  float h2 = bf2f((unsigned short)(q1 & 0xffffu));
  float h3 = bf2f((unsigned short)(q1 >> 16));
  float* dst = agg + d * DP + c0;
  atomAddF(dst + 0, h0 + p1[0] + p2[0]);
  atomAddF(dst + 1, h1 + p1[1] + p2[1]);
  atomAddF(dst + 2, h2 + p1[2] + p2[2]);
  atomAddF(dst + 3, h3 + p1[3] + p2[3]);
}

// per-column sum / sumsq over 512-row chunks
__global__ void k_bn_stats(const float* __restrict__ h, float* __restrict__ sums,
                           float* __restrict__ sq) {
  int c  = threadIdx.x;                 // 0..319
  int r0 = blockIdx.x * 512;
  float s = 0.f, q = 0.f;
  for (int r = 0; r < 512; ++r) {
    float v = h[(r0 + r) * DP + c];
    s += v; q += v * v;
  }
  atomAddF(&sums[c], s);
  atomAddF(&sq[c], q);
}

__global__ void k_bn_apply(const float* __restrict__ h,
                           const float* __restrict__ sums, const float* __restrict__ sq,
                           const float* __restrict__ gamma, const float* __restrict__ beta,
                           unsigned* __restrict__ hbf, int do_relu) {
  int i = blockIdx.x * blockDim.x + threadIdx.x;   // NN * DP/2
  if (i >= NN * (DP / 2)) return;
  int j = i % (DP / 2), c0 = j * 2;
  if (c0 >= DD) { hbf[i] = 0u; return; }
  const float invN = 1.f / (float)NN;
  float mu0 = sums[c0] * invN,     mu1 = sums[c0 + 1] * invN;
  float va0 = sq[c0] * invN - mu0 * mu0;
  float va1 = sq[c0 + 1] * invN - mu1 * mu1;
  float2 v = reinterpret_cast<const float2*>(h)[i];
  float o0 = (v.x - mu0) * rsqrtf(va0 + BNEPS) * gamma[c0]     + beta[c0];
  float o1 = (v.y - mu1) * rsqrtf(va1 + BNEPS) * gamma[c0 + 1] + beta[c0 + 1];
  if (do_relu) { o0 = fmaxf(o0, 0.f); o1 = fmaxf(o1, 0.f); }
  hbf[i] = pack2(o0, o1);
}

__global__ void k_pool(const unsigned* __restrict__ hbf, const int* __restrict__ batch,
                       float* __restrict__ pooled, float* __restrict__ counts) {
  int i = blockIdx.x * blockDim.x + threadIdx.x;   // NN * DP/2
  if (i >= NN * (DP / 2)) return;
  int n = i / (DP / 2), j = i % (DP / 2), c0 = j * 2;
  int g = batch[n];
  unsigned p = hbf[i];
  atomAddF(&pooled[g * DP + c0],     bf2f((unsigned short)(p & 0xffffu)));
  atomAddF(&pooled[g * DP + c0 + 1], bf2f((unsigned short)(p >> 16)));
  if (j == 0) atomAddF(&counts[g], 1.f);
}

__global__ void k_hg(const float* __restrict__ pooled, const float* __restrict__ counts,
                     unsigned* __restrict__ hgbf) {
  int i = blockIdx.x * blockDim.x + threadIdx.x;   // GG * DP/2
  if (i >= GG * (DP / 2)) return;
  int g = i / (DP / 2);
  float inv = 1.f / fmaxf(counts[g], 1.f);
  float2 p = reinterpret_cast<const float2*>(pooled)[i];
  hgbf[i] = pack2(p.x * inv, p.y * inv);
}

__global__ void k_concat(const float* __restrict__ hg, const float* __restrict__ tx,
                         unsigned short* __restrict__ zin) {
  int i = blockIdx.x * blockDim.x + threadIdx.x;   // GG * CZ
  if (i >= GG * CZ) return;
  int g = i / CZ, j = i % CZ;
  float v = (j < FEATN) ? hg[g * FEATN + j] : tx[g * EXN + (j - FEATN)];
  zin[i] = f2bf(v);
}

__global__ void k_pred3(const unsigned short* __restrict__ z,
                        const float* __restrict__ W, const float* __restrict__ b,
                        float* __restrict__ out) {
  int i = blockIdx.x * blockDim.x + threadIdx.x;   // GG * 2
  if (i >= GG * 2) return;
  int g = i / 2, o = i & 1;
  float acc = b[o];
  for (int k = 0; k < PDN; ++k)
    acc += bf2f(z[g * PDN + k]) * W[k * 2 + o];
  out[i] = acc;
}

// ---------------------------------------------------------------------------

extern "C" void kernel_launch(void* const* d_in, const int* in_sizes, int n_in,
                              void* d_out, int out_size, void* d_ws, size_t ws_size,
                              hipStream_t stream) {
  (void)in_sizes; (void)n_in; (void)out_size; (void)ws_size;

  const int*   x        = (const int*)d_in[0];
  const int*   ei       = (const int*)d_in[1];
  const int*   ea       = (const int*)d_in[2];
  const int*   batch    = (const int*)d_in[3];
  const float* text_x   = (const float*)d_in[4];
  const float* x_emb1   = (const float*)d_in[7];
  const float* x_emb2   = (const float*)d_in[8];
  const float* gin_e1   = (const float*)d_in[9];
  const float* gin_e2   = (const float*)d_in[10];
  const float* gin_W1   = (const float*)d_in[11];
  const float* gin_b1   = (const float*)d_in[12];
  const float* gin_W2   = (const float*)d_in[13];
  const float* gin_b2   = (const float*)d_in[14];
  const float* bn_g     = (const float*)d_in[15];
  const float* bn_b     = (const float*)d_in[16];
  const float* feat_W   = (const float*)d_in[17];
  const float* feat_b   = (const float*)d_in[18];
  const float* text_W   = (const float*)d_in[19];
  const float* text_b   = (const float*)d_in[20];
  const float* pred_W1  = (const float*)d_in[21];
  const float* pred_b1  = (const float*)d_in[22];
  const float* pred_W2  = (const float*)d_in[23];
  const float* pred_b2  = (const float*)d_in[24];
  const float* pred_W3  = (const float*)d_in[25];
  const float* pred_b3  = (const float*)d_in[26];

  char* w = (char*)d_ws;
  auto carve = [&](size_t bytes) -> void* {
    void* p = (void*)w;
    w += (bytes + 255) & ~(size_t)255;
    return p;
  };
  unsigned short* hbf   = (unsigned short*)carve((size_t)NN * DP * 2);
  float*          agg   = (float*)         carve((size_t)NN * DP * 4);
  unsigned short* hid   = (unsigned short*)carve((size_t)NN * HP * 2);
  unsigned short* wt1   = (unsigned short*)carve((size_t)LL * HP * DP * 2); // [l][640][320]
  unsigned short* wt2   = (unsigned short*)carve((size_t)LL * DP * HP * 2); // [l][320][640]
  float*          b1p   = (float*)         carve((size_t)LL * HP * 4);
  float*          b2p   = (float*)         carve((size_t)LL * DP * 4);
  float*          bns   = (float*)         carve((size_t)2 * DP * 4);       // sums | sumsq
  float*          pooled= (float*)         carve((size_t)GG * DP * 4 + GG * 4); // + counts
  float*          counts= pooled + (size_t)GG * DP;
  unsigned short* hgbf  = (unsigned short*)carve((size_t)GG * DP * 2);
  unsigned short* fWt   = (unsigned short*)carve((size_t)FEATN * DP * 2);
  unsigned short* tWt   = (unsigned short*)carve((size_t)EXN * TIN * 2);
  unsigned short* txbf  = (unsigned short*)carve((size_t)GG * TIN * 2);
  float*          tx    = (float*)         carve((size_t)GG * EXN * 4);
  unsigned short* zin   = (unsigned short*)carve((size_t)GG * CZ * 2);
  unsigned short* pw1t  = (unsigned short*)carve((size_t)PDN * CZ * 2);
  unsigned short* pw2t  = (unsigned short*)carve((size_t)PDN * PDN * 2);
  unsigned short* z1    = (unsigned short*)carve((size_t)GG * PDN * 2);
  unsigned short* z2    = (unsigned short*)carve((size_t)GG * PDN * 2);

  const int T = 256;
  auto blks = [](int n) { return (n + 255) / 256; };

  // --- weight / bias preprocessing (bf16, transposed, padded) ---
  for (int l = 0; l < LL; ++l) {
    k_wconv_t<<<blks(HP * DP), T, 0, stream>>>(gin_W1 + (size_t)l * DD * HH,
                                               wt1 + (size_t)l * HP * DP, DD, HH, DP, HP);
    k_wconv_t<<<blks(DP * HP), T, 0, stream>>>(gin_W2 + (size_t)l * HH * DD,
                                               wt2 + (size_t)l * DP * HP, HH, DD, HP, DP);
    k_biaspad<<<blks(HP), T, 0, stream>>>(gin_b1 + (size_t)l * HH, b1p + (size_t)l * HP, HH, HP);
    k_biaspad<<<blks(DP), T, 0, stream>>>(gin_b2 + (size_t)l * DD, b2p + (size_t)l * DP, DD, DP);
  }
  k_wconv_t<<<blks(FEATN * DP), T, 0, stream>>>(feat_W, fWt, DD, FEATN, DP, FEATN);
  k_wconv_t<<<blks(EXN * TIN), T, 0, stream>>>(text_W, tWt, TIN, EXN, TIN, EXN);
  k_wconv_t<<<blks(PDN * CZ), T, 0, stream>>>(pred_W1, pw1t, CZ, PDN, CZ, PDN);
  k_wconv_t<<<blks(PDN * PDN), T, 0, stream>>>(pred_W2, pw2t, PDN, PDN, PDN, PDN);

  // --- node embedding ---
  const int NPAIR = NN * (DP / 2);
  k_embed<<<blks(NPAIR), T, 0, stream>>>(x, x_emb1, x_emb2, (unsigned*)hbf);

  // --- GIN layers ---
  for (int l = 0; l < LL; ++l) {
    k_agg_init<<<blks(NPAIR), T, 0, stream>>>((const unsigned*)hbf,
                                              gin_e1 + (size_t)(l * 5 + 4) * DD,
                                              gin_e2 + (size_t)(l * 3 + 0) * DD, agg);
    k_edge<<<blks(EE * 75), T, 0, stream>>>(ei, ea, (const unsigned*)hbf,
                                            gin_e1 + (size_t)l * 5 * DD,
                                            gin_e2 + (size_t)l * 3 * DD, agg);
    k_cvt_pairs<<<blks(NPAIR), T, 0, stream>>>(agg, (unsigned*)hbf, NPAIR);
    // hid = relu(agg @ W1 + b1)  [N,640] bf16
    k_gemm<1><<<dim3(HP / 32, NN / 256), T, 0, stream>>>(hbf, DP, wt1 + (size_t)l * HP * DP, DP,
                                                         b1p + (size_t)l * HP, nullptr, hid, HP, DP);
    // h_f32 = hid @ W2 + b2 -> agg buffer reused
    k_gemm<0><<<dim3(DP / 32, NN / 256), T, 0, stream>>>(hid, HP, wt2 + (size_t)l * DP * HP, HP,
                                                         b2p + (size_t)l * DP, agg, nullptr, DP, HP);
    k_zero<<<blks(2 * DP), T, 0, stream>>>(bns, 2 * DP);
    k_bn_stats<<<NN / 512, DP, 0, stream>>>(agg, bns, bns + DP);
    k_bn_apply<<<blks(NPAIR), T, 0, stream>>>(agg, bns, bns + DP,
                                              bn_g + (size_t)l * DD, bn_b + (size_t)l * DD,
                                              (unsigned*)hbf, (l < LL - 1) ? 1 : 0);
  }

  // --- global mean pool ---
  k_zero<<<blks(GG * DP + GG), T, 0, stream>>>(pooled, GG * DP + GG);
  k_pool<<<blks(NPAIR), T, 0, stream>>>((const unsigned*)hbf, batch, pooled, counts);
  k_hg<<<blks(GG * (DP / 2)), T, 0, stream>>>(pooled, counts, (unsigned*)hgbf);

  // --- heads ---
  float* hg_out   = (float*)d_out;                 // [G, 512]
  float* pred_out = (float*)d_out + (size_t)GG * FEATN;
  k_gemm<0><<<dim3(FEATN / 32, GG / 256), T, 0, stream>>>(hgbf, DP, fWt, DP, feat_b,
                                                          hg_out, nullptr, FEATN, DP);
  k_cvt_pairs<<<blks(GG * TIN / 2), T, 0, stream>>>(text_x, (unsigned*)txbf, GG * TIN / 2);
  k_gemm<0><<<dim3(EXN / 32, GG / 256), T, 0, stream>>>(txbf, TIN, tWt, TIN, text_b,
                                                        tx, nullptr, EXN, TIN);
  k_concat<<<blks(GG * CZ), T, 0, stream>>>(hg_out, tx, zin);
  k_gemm<2><<<dim3(PDN / 32, GG / 256), T, 0, stream>>>(zin, CZ, pw1t, CZ, pred_b1,
                                                        nullptr, z1, PDN, CZ);
  k_gemm<2><<<dim3(PDN / 32, GG / 256), T, 0, stream>>>(z1, PDN, pw2t, PDN, pred_b2,
                                                        nullptr, z2, PDN, PDN);
  k_pred3<<<blks(GG * 2), T, 0, stream>>>(z2, pred_W3, pred_b3, pred_out);
}
